// MultiHeadAttention_36000415875247
// MI455X (gfx1250) — matrix operand order repaired and checked by reading, not compiled, not run on hardware
//
#include <hip/hip_runtime.h>
#include <hip/hip_bf16.h>

// ---------------- problem constants ----------------
constexpr int Bsz = 2;
constexpr int Ssz = 2048;
constexpr int Hsz = 1024;
constexpr int NHn = 16;
constexpr int HDn = 64;   // head dim

typedef __attribute__((ext_vector_type(16))) _Float16 v16h;
typedef __attribute__((ext_vector_type(8)))  _Float16 v8h;
typedef __attribute__((ext_vector_type(8)))  float    v8f;

// ---------------- WMMA fragment helpers (layouts per CDNA5 ISA §7.12.2) ----------------
__device__ __forceinline__ v16h make_frag(v8h lo, v8h hi) {
  v16h r;
#pragma unroll
  for (int i = 0; i < 8; ++i) { r[i] = lo[i]; r[i + 8] = hi[i]; }
  return r;
}

// A fragment, 16x32 f16 (MxK). lane m = lane&15, h = lane>>4.
// halves 0..7  -> K = h*8 + j ; halves 8..15 -> K = 16 + h*8 + (j-8)
__device__ __forceinline__ v16h load_a16(const _Float16* rowp, int h) {
  v8h lo = *(const v8h*)(rowp + h * 8);
  v8h hi = *(const v8h*)(rowp + 16 + h * 8);
  return make_frag(lo, hi);
}

// B fragment, 32x16 f16 (KxN). lane col = lane&15, h = lane>>4.
// halves 0..7  -> K = h*8 + j ; halves 8..15 -> K = 16 + h*8 + (j-8)
__device__ __forceinline__ v16h load_b16(const _Float16* colp, int h) {
  v8h lo = *(const v8h*)(colp + h * 8);
  v8h hi = *(const v8h*)(colp + 16 + h * 8);
  return make_frag(lo, hi);
}

__device__ __forceinline__ v8f wmma_f16(v16h a, v16h b, v8f c) {
  return __builtin_amdgcn_wmma_f32_16x16x32_f16(false, a, false, b, (short)0, c, false, false);
}

// ---------------- f32 -> f16 convert ----------------
__global__ __launch_bounds__(256) void cvt_f32_f16(const float* __restrict__ src,
                                                   _Float16* __restrict__ dst, int n) {
  int i = blockIdx.x * blockDim.x + threadIdx.x;
  int stride = gridDim.x * blockDim.x;
  for (; i < n; i += stride) dst[i] = (_Float16)src[i];
}

// ---------------- GEMM: Y = A(M x 1024) @ W^T + bias ----------------
// Wave tile 32(M) x 64(N): 2 A-frags x 4 B-frags -> 8 WMMAs per 32-wide K chunk.
// Fragments for chunk k+1 are prefetched into separate registers before the
// chunk-k WMMAs issue, so global loads overlap the matrix pipe.
// mode 0: f16 output scattered to [b, nh, s, hd] (QKV projections)
// mode 1: f32 output row-major [M, H] (output projection -> d_out)
__global__ __launch_bounds__(256) void gemm_h1024(const _Float16* __restrict__ A,
                                                  const _Float16* __restrict__ W,
                                                  const float* __restrict__ bias,
                                                  _Float16* __restrict__ out_h,
                                                  float* __restrict__ out_f,
                                                  int mode) {
  const int lane = threadIdx.x & 31;
  const int wave = threadIdx.x >> 5;
  const int wid  = blockIdx.x * 8 + wave;
  const int NT   = Hsz / 64;           // 16 n-tiles of width 64
  const int mt = wid / NT, nt = wid % NT;
  const int m0 = mt * 32, n0 = nt * 64;
  const int h   = lane >> 4;
  const int col = lane & 15;

  const _Float16* arow0 = A + (size_t)(m0 + col) * Hsz;
  const _Float16* arow1 = arow0 + (size_t)16 * Hsz;
  const _Float16* wr0 = W + (size_t)(n0 + col) * Hsz;
  const _Float16* wr1 = wr0 + (size_t)16 * Hsz;
  const _Float16* wr2 = wr0 + (size_t)32 * Hsz;
  const _Float16* wr3 = wr0 + (size_t)48 * Hsz;

  v8f acc[2][4] = {};

  // prologue: fragments for k = 0
  v16h a0 = load_a16(arow0, h);
  v16h a1 = load_a16(arow1, h);
  v16h b0 = load_b16(wr0, h);
  v16h b1 = load_b16(wr1, h);
  v16h b2 = load_b16(wr2, h);
  v16h b3 = load_b16(wr3, h);

  for (int k = 32; k < Hsz; k += 32) {
    // prefetch next chunk into fresh registers
    v16h a0n = load_a16(arow0 + k, h);
    v16h a1n = load_a16(arow1 + k, h);
    v16h b0n = load_b16(wr0 + k, h);
    v16h b1n = load_b16(wr1 + k, h);
    v16h b2n = load_b16(wr2 + k, h);
    v16h b3n = load_b16(wr3 + k, h);
    // current chunk WMMAs (overlap with the loads above)
    acc[0][0] = wmma_f16(a0, b0, acc[0][0]);
    acc[0][1] = wmma_f16(a0, b1, acc[0][1]);
    acc[0][2] = wmma_f16(a0, b2, acc[0][2]);
    acc[0][3] = wmma_f16(a0, b3, acc[0][3]);
    acc[1][0] = wmma_f16(a1, b0, acc[1][0]);
    acc[1][1] = wmma_f16(a1, b1, acc[1][1]);
    acc[1][2] = wmma_f16(a1, b2, acc[1][2]);
    acc[1][3] = wmma_f16(a1, b3, acc[1][3]);
    a0 = a0n; a1 = a1n; b0 = b0n; b1 = b1n; b2 = b2n; b3 = b3n;
  }
  // epilogue chunk
  acc[0][0] = wmma_f16(a0, b0, acc[0][0]);
  acc[0][1] = wmma_f16(a0, b1, acc[0][1]);
  acc[0][2] = wmma_f16(a0, b2, acc[0][2]);
  acc[0][3] = wmma_f16(a0, b3, acc[0][3]);
  acc[1][0] = wmma_f16(a1, b0, acc[1][0]);
  acc[1][1] = wmma_f16(a1, b1, acc[1][1]);
  acc[1][2] = wmma_f16(a1, b2, acc[1][2]);
  acc[1][3] = wmma_f16(a1, b3, acc[1][3]);

#pragma unroll
  for (int mi = 0; mi < 2; ++mi) {
#pragma unroll
    for (int t = 0; t < 4; ++t) {
      const int n = n0 + t * 16 + col;
      const float bval = bias[n];
#pragma unroll
      for (int r = 0; r < 8; ++r) {
        const int row = m0 + mi * 16 + r + h * 8;  // C layout: lanes 16-31 hold M=r+8
        const float v = acc[mi][t][r] + bval;
        if (mode == 0) {
          const int bb = row / Ssz, ss = row % Ssz;
          const int nh = n >> 6, hd = n & 63;
          out_h[(((size_t)bb * NHn + nh) * Ssz + ss) * HDn + hd] = (_Float16)v;
        } else {
          out_f[(size_t)row * Hsz + n] = v;
        }
      }
    }
  }
}

// ---------------- flash attention ----------------
// Block = 256 threads = 8 waves; each wave owns a 16-row Q tile (128 q rows / block).
// K block staged with GLOBAL_LOAD_ASYNC_TO_LDS_B128 (ASYNCcnt), V transposed
// through VGPRs, online softmax with shfl_xor row reductions, P transposed via LDS.
__global__ __launch_bounds__(256) void flash_attn(const _Float16* __restrict__ Qh,
                                                  const _Float16* __restrict__ Kh,
                                                  const _Float16* __restrict__ Vh,
                                                  const int* __restrict__ mask,
                                                  _Float16* __restrict__ Ctx) {
  __shared__ alignas(16) _Float16 Ks[32][64];     // [key][d]
  __shared__ alignas(16) _Float16 VsT[64][32];    // [d][key]  (transposed)
  __shared__ alignas(16) _Float16 Ps[8][16][32];  // per-wave P tile [row][key]

  const int lane = threadIdx.x & 31;
  const int wave = threadIdx.x >> 5;
  const int h    = lane >> 4;
  const int col  = lane & 15;

  const int QB = 128;
  const int bh = blockIdx.x / (Ssz / QB);
  const int qb = blockIdx.x % (Ssz / QB);
  const int b  = bh / NHn;
  const int head = bh % NHn;
  const int q0 = qb * QB + wave * 16;

  const _Float16* Qbase = Qh + ((size_t)bh * Ssz + q0) * HDn;
  const _Float16* Kbase = Kh + (size_t)bh * Ssz * HDn;
  const _Float16* Vbase = Vh + (size_t)bh * Ssz * HDn;
  const int* mrow = mask + (size_t)b * Ssz;

  // Q fragments for d 0-31 and d 32-63
  const _Float16* qrow = Qbase + (size_t)(lane & 15) * HDn;
  const v16h qa0 = load_a16(qrow, h);
  const v16h qa1 = load_a16(qrow + 32, h);

  v8f o[4] = {};
  float mr[8], lr[8];
#pragma unroll
  for (int r = 0; r < 8; ++r) { mr[r] = -1e30f; lr[r] = 0.f; }

  // per-thread staging coordinates: 256 threads x 16B for K (async) and V (transpose)
  const int krow  = threadIdx.x >> 3;
  const int chunk = threadIdx.x & 7;
  // low 32 bits of a flat shared-memory address are the LDS byte address (ISA §10.2)
  const unsigned lds_k = (unsigned)(uintptr_t)&Ks[krow][chunk * 8];

  for (int kb = 0; kb < Ssz; kb += 32) {
    __syncthreads();  // previous iteration's LDS reads done
    {
      // K block: async copy global -> LDS, 16B per thread, no VGPR round-trip
      const unsigned long long gk =
          (unsigned long long)(uintptr_t)(Kbase + (size_t)(kb + krow) * HDn + chunk * 8);
      asm volatile("global_load_async_to_lds_b128 %0, %1, off"
                   :: "v"(lds_k), "v"(gk) : "memory");
      // V block: load 8 halves and scatter transposed
      v8h vv = *(const v8h*)(Vbase + (size_t)(kb + krow) * HDn + chunk * 8);
#pragma unroll
      for (int j = 0; j < 8; ++j) VsT[chunk * 8 + j][krow] = vv[j];
      asm volatile("s_wait_asynccnt 0" ::: "memory");
    }
    __syncthreads();  // staging visible to all waves

    // scores: two 16x16 tiles (keys kb..kb+15, kb+16..kb+31), contraction over d=64
    v8f c0 = {}, c1 = {};
    c0 = wmma_f16(qa0, load_b16(&Ks[col][0], h), c0);
    c0 = wmma_f16(qa1, load_b16(&Ks[col][32], h), c0);
    c1 = wmma_f16(qa0, load_b16(&Ks[16 + col][0], h), c1);
    c1 = wmma_f16(qa1, load_b16(&Ks[16 + col][32], h), c1);

    const float add0 = (mrow[kb + col] == 0) ? -1e9f : 0.f;
    const float add1 = (mrow[kb + 16 + col] == 0) ? -1e9f : 0.f;

    float s0[8], s1[8], p0[8], p1[8];
#pragma unroll
    for (int r = 0; r < 8; ++r) {
      s0[r] = c0[r] * 0.125f + add0;   // 1/sqrt(64)
      s1[r] = c1[r] * 0.125f + add1;
      // row max across the 16 key columns (lanes within each 16-lane half)
      float v = fmaxf(s0[r], s1[r]);
#pragma unroll
      for (int mskw = 1; mskw < 16; mskw <<= 1) v = fmaxf(v, __shfl_xor(v, mskw, 32));
      const float mnew = fmaxf(mr[r], v);
      p0[r] = __expf(s0[r] - mnew);
      p1[r] = __expf(s1[r] - mnew);
      float rs = p0[r] + p1[r];
#pragma unroll
      for (int mskw = 1; mskw < 16; mskw <<= 1) rs += __shfl_xor(rs, mskw, 32);
      const float alpha = __expf(mr[r] - mnew);
      lr[r] = lr[r] * alpha + rs;
      mr[r] = mnew;
#pragma unroll
      for (int t = 0; t < 4; ++t) o[t][r] *= alpha;
      // spill P (C-layout) to LDS for transpose into A-layout
      const int row = r + h * 8;
      Ps[wave][row][col]      = (_Float16)p0[r];
      Ps[wave][row][16 + col] = (_Float16)p1[r];
    }

    // intra-wave LDS RAW fence (different lanes wrote the rows we now read)
    asm volatile("s_wait_dscnt 0" ::: "memory");

    const v16h pa = load_a16(&Ps[wave][lane & 15][0], h);  // 16x32 P fragment
#pragma unroll
    for (int t = 0; t < 4; ++t) {
      v16h vb = load_b16(&VsT[t * 16 + col][0], h);        // 32(key) x 16(hd)
      o[t] = wmma_f16(pa, vb, o[t]);
    }
  }

  // normalize and write context in [b, s, nh*64+hd] layout for the output GEMM
#pragma unroll
  for (int r = 0; r < 8; ++r) {
    const float inv = 1.f / lr[r];
    const int row = r + h * 8;
#pragma unroll
    for (int t = 0; t < 4; ++t) {
      const int hd = t * 16 + col;
      Ctx[((size_t)b * Ssz + q0 + row) * Hsz + head * HDn + hd] =
          (_Float16)(o[t][r] * inv);
    }
  }
}

// ---------------- host launch ----------------
extern "C" void kernel_launch(void* const* d_in, const int* in_sizes, int n_in,
                              void* d_out, int out_size, void* d_ws, size_t ws_size,
                              hipStream_t stream) {
  const float* hs  = (const float*)d_in[0];
  const int*  mask = (const int*)d_in[1];
  const float* Wq = (const float*)d_in[2];
  const float* bq = (const float*)d_in[3];
  const float* Wk = (const float*)d_in[4];
  const float* bk = (const float*)d_in[5];
  const float* Wv = (const float*)d_in[6];
  const float* bv = (const float*)d_in[7];
  const float* Wo = (const float*)d_in[8];
  const float* bo = (const float*)d_in[9];
  float* out = (float*)d_out;

  const size_t MH = (size_t)Bsz * Ssz * Hsz;  // 4M elements
  const size_t WH = (size_t)Hsz * Hsz;        // 1M elements
  _Float16* Xh  = (_Float16*)d_ws;
  _Float16* Wqh = Xh + MH;
  _Float16* Wkh = Wqh + WH;
  _Float16* Wvh = Wkh + WH;
  _Float16* Woh = Wvh + WH;
  _Float16* Qh  = Woh + WH;
  _Float16* Kh  = Qh + MH;
  _Float16* Vh  = Kh + MH;
  _Float16* Ch  = Vh + MH;   // total 24M halves = 48 MB of workspace

  // converts (grid-stride)
  cvt_f32_f16<<<2048, 256, 0, stream>>>(hs, Xh, (int)MH);
  cvt_f32_f16<<<1024, 256, 0, stream>>>(Wq, Wqh, (int)WH);
  cvt_f32_f16<<<1024, 256, 0, stream>>>(Wk, Wkh, (int)WH);
  cvt_f32_f16<<<1024, 256, 0, stream>>>(Wv, Wvh, (int)WH);
  cvt_f32_f16<<<1024, 256, 0, stream>>>(Wo, Woh, (int)WH);

  // GEMMs: (B*S/32) m-tiles * (H/64) n-tiles = 2048 wave-tiles / 8 waves = 256 blocks
  const int gemm_blocks = (Bsz * Ssz / 32) * (Hsz / 64) / 8;
  gemm_h1024<<<gemm_blocks, 256, 0, stream>>>(Xh, Wqh, bq, Qh, nullptr, 0);
  gemm_h1024<<<gemm_blocks, 256, 0, stream>>>(Xh, Wkh, bk, Kh, nullptr, 0);
  gemm_h1024<<<gemm_blocks, 256, 0, stream>>>(Xh, Wvh, bv, Vh, nullptr, 0);

  // attention: (B*NH) heads * (S/128) q-blocks = 512 blocks
  flash_attn<<<(Bsz * NHn) * (Ssz / 128), 256, 0, stream>>>(Qh, Kh, Vh, mask, Ch);

  // output projection -> f32 d_out
  gemm_h1024<<<gemm_blocks, 256, 0, stream>>>(Ch, Woh, bo, nullptr, out, 1);
}
